// MultiHeadSelfAttention_15857019257437
// MI455X (gfx1250) — matrix operand order repaired and checked by reading, not compiled, not run on hardware
//
#include <hip/hip_runtime.h>
#include <hip/hip_bf16.h>
#include <math.h>

// ---------------------------------------------------------------------------
// Multi-head self-attention for MI455X (gfx1250, wave32, WMMA).
// All three contraction stages (QKV proj, QK^T / PV, out proj) run on
// v_wmma_f32_16x16x32_f16 with f32 accumulation; f32 inputs are converted to
// f16 once while staging through LDS.
// ---------------------------------------------------------------------------

#define BS  4
#define SEQ 2048
#define DIM 1024
#define NH  16
#define DH  64

typedef __attribute__((ext_vector_type(16))) _Float16 v16h;
typedef __attribute__((ext_vector_type(8)))  _Float16 v8h;
typedef __attribute__((ext_vector_type(8)))  float    v8f;
typedef __attribute__((ext_vector_type(4)))  float    v4f;

__device__ __forceinline__ v8f wmma_f16(v16h a, v16h b, v8f c) {
    // (neg_a, A, neg_b, B, c_mod, C, reuse_a, reuse_b)
    return __builtin_amdgcn_wmma_f32_16x16x32_f16(false, a, false, b, (short)0, c,
                                                  false, false);
}

// A fragment (16x32, f16) from LDS. ISA 7.12.2 layout: lane L holds row
// m0 + (L&15); lanes 0-15 hold K {0..7, 16..23}, lanes 16-31 {8..15, 24..31}.
__device__ __forceinline__ v16h load_frag_a(const _Float16* base, int ld,
                                            int m0, int k0, int lane) {
    const _Float16* rp = base + (m0 + (lane & 15)) * ld + k0 + ((lane >> 4) << 3);
    union { v16h v; v8h h[2]; } u;
    u.h[0] = *(const v8h*)(rp);
    u.h[1] = *(const v8h*)(rp + 16);
    return u.v;
}

// B fragment (32x16, f16) from LDS stored as [n][k] (k contiguous). Lane L
// holds column n0 + (L&15); lanes 0-15 K {0..7, 16..23}, lanes 16-31 {8..15, 24..31}.
__device__ __forceinline__ v16h load_frag_b(const _Float16* base, int ld,
                                            int n0, int k0, int lane) {
    const _Float16* cp = base + (n0 + (lane & 15)) * ld + k0 + ((lane >> 4) << 3);
    union { v16h v; v8h h[2]; } u;
    u.h[0] = *(const v8h*)(cp);
    u.h[1] = *(const v8h*)(cp + 16);
    return u.v;
}

__device__ __forceinline__ v8h cvt8(v4f a, v4f b) {
    v8h r;
    r[0] = (_Float16)a.x; r[1] = (_Float16)a.y; r[2] = (_Float16)a.z; r[3] = (_Float16)a.w;
    r[4] = (_Float16)b.x; r[5] = (_Float16)b.y; r[6] = (_Float16)b.z; r[7] = (_Float16)b.w;
    return r;
}

// ---------------------------------------------------------------------------
// Kernel 1: projection  Out = (X @ W^T + bias) * scale, stored f16 in
// per-head layout Out[((b*NH + h)*SEQ + s)*DH + d].  Block tile 128x64, K=64.
// ---------------------------------------------------------------------------
__global__ __launch_bounds__(256)
void mhsa_proj_kernel(const float* __restrict__ X, const float* __restrict__ W,
                      const float* __restrict__ bias, _Float16* __restrict__ Out,
                      float scale) {
    __shared__ _Float16 As[128][72];
    __shared__ _Float16 Bs[64][72];

    const int tid  = threadIdx.x;
    const int lane = tid & 31;
    const int w    = tid >> 5;
    const int wm   = w >> 1;      // 0..3  -> 32-row slice
    const int wn   = w & 1;       // 0..1  -> 32-col slice
    const int m_block = blockIdx.x * 128;
    const int n_block = blockIdx.y * 64;

    v8f acc[2][2] = {};

    const int ar = tid >> 1, ac = (tid & 1) * 32;   // A stage: 32 f32 each
    const int br = tid >> 2, bc = (tid & 3) * 16;   // B stage: 16 f32 each

    for (int k0 = 0; k0 < DIM; k0 += 64) {
        __syncthreads();
        {   // stage X tile (f32 -> f16)
            const float* gp = X + (size_t)(m_block + ar) * DIM + k0 + ac;
#pragma unroll
            for (int i = 0; i < 4; ++i) {
                v4f f0 = *(const v4f*)(gp + 8 * i);
                v4f f1 = *(const v4f*)(gp + 8 * i + 4);
                *(v8h*)&As[ar][ac + 8 * i] = cvt8(f0, f1);
            }
        }
        {   // stage W tile (rows of W give contiguous K: B[n][k] = W[n][k])
            const float* gw = W + (size_t)(n_block + br) * DIM + k0 + bc;
#pragma unroll
            for (int i = 0; i < 2; ++i) {
                v4f f0 = *(const v4f*)(gw + 8 * i);
                v4f f1 = *(const v4f*)(gw + 8 * i + 4);
                *(v8h*)&Bs[br][bc + 8 * i] = cvt8(f0, f1);
            }
        }
        __syncthreads();
#pragma unroll
        for (int kk = 0; kk < 64; kk += 32) {
            v16h a0 = load_frag_a(&As[0][0], 72, 32 * wm,      kk, lane);
            v16h a1 = load_frag_a(&As[0][0], 72, 32 * wm + 16, kk, lane);
            v16h b0 = load_frag_b(&Bs[0][0], 72, 32 * wn,      kk, lane);
            v16h b1 = load_frag_b(&Bs[0][0], 72, 32 * wn + 16, kk, lane);
            acc[0][0] = wmma_f16(a0, b0, acc[0][0]);
            acc[0][1] = wmma_f16(a0, b1, acc[0][1]);
            acc[1][0] = wmma_f16(a1, b0, acc[1][0]);
            acc[1][1] = wmma_f16(a1, b1, acc[1][1]);
        }
    }

    // epilogue: +bias, *scale, scatter into per-head layout as f16
#pragma unroll
    for (int mi = 0; mi < 2; ++mi)
#pragma unroll
        for (int ni = 0; ni < 2; ++ni)
#pragma unroll
            for (int r = 0; r < 8; ++r) {
                int m = m_block + 32 * wm + 16 * mi + r + 8 * (lane >> 4);
                int n = n_block + 32 * wn + 16 * ni + (lane & 15);
                int bidx = m >> 11, s = m & (SEQ - 1);
                int h = n >> 6, d = n & (DH - 1);
                float val = (acc[mi][ni][r] + bias[n]) * scale;
                Out[(((size_t)(bidx * NH + h)) * SEQ + s) * DH + d] = (_Float16)val;
            }
}

// ---------------------------------------------------------------------------
// Kernel 2: flash attention per (b,h).  64-query tile per block, key tiles of
// 64, online softmax.  Scores and context both via WMMA.
// ---------------------------------------------------------------------------
__global__ __launch_bounds__(256)
void mhsa_attn_kernel(const _Float16* __restrict__ Qh, const _Float16* __restrict__ Kh,
                      const _Float16* __restrict__ Vh, const int* __restrict__ mask,
                      _Float16* __restrict__ Ctx) {
    __shared__ _Float16 Qs[64][72];
    __shared__ _Float16 Ks[64][72];
    __shared__ _Float16 Vt[64][72];   // transposed V tile: Vt[d][key]
    __shared__ _Float16 Ps[64][72];   // probabilities
    __shared__ float    Sc[64][68];   // raw scores
    __shared__ float    rmax[64], rsum[64], rscl[64];
    __shared__ int      smask[64];

    const int tid  = threadIdx.x;
    const int lane = tid & 31;
    const int w    = tid >> 5;
    const int wm   = w >> 1;          // 16-row slice of the 64-row q tile
    const int wn   = w & 1;           // 32-col slice
    const int qt = blockIdx.x, h = blockIdx.y, b = blockIdx.z;

    const size_t base = ((size_t)(b * NH + h)) * SEQ * DH;
    const _Float16* Qp = Qh + base + (size_t)qt * 64 * DH;
    const _Float16* Kp = Kh + base;
    const _Float16* Vp = Vh + base;
    const int* mp = mask + b * SEQ;

    {   // load Q tile (resident for whole kernel)
        int r = tid >> 2, c = (tid & 3) * 16;
        const v8h* g = (const v8h*)(Qp + r * DH + c);
        *(v8h*)&Qs[r][c]     = g[0];
        *(v8h*)&Qs[r][c + 8] = g[1];
    }
    if (tid < 64) { rmax[tid] = -3.0e38f; rsum[tid] = 0.0f; }

    v8f o[2] = {};

    for (int kt = 0; kt < SEQ / 64; ++kt) {
        __syncthreads();
        {   // stage K tile and transposed V tile
            int r = tid >> 2, c = (tid & 3) * 16;
            const v8h* gk = (const v8h*)(Kp + (size_t)(kt * 64 + r) * DH + c);
            *(v8h*)&Ks[r][c]     = gk[0];
            *(v8h*)&Ks[r][c + 8] = gk[1];
            const v8h* gv = (const v8h*)(Vp + (size_t)(kt * 64 + r) * DH + c);
            v8h v0 = gv[0], v1 = gv[1];
#pragma unroll
            for (int i = 0; i < 8; ++i) {
                Vt[c + i][r]     = v0[i];
                Vt[c + 8 + i][r] = v1[i];
            }
            if (tid < 64) smask[tid] = mp[kt * 64 + tid];
        }
        __syncthreads();

        // scores = Q_tile @ K_tile^T   (contract over d)
        v8f sc[2] = {};
#pragma unroll
        for (int kk = 0; kk < DH; kk += 32) {
            v16h a = load_frag_a(&Qs[0][0], 72, 16 * wm, kk, lane);
#pragma unroll
            for (int ni = 0; ni < 2; ++ni) {
                v16h bf = load_frag_b(&Ks[0][0], 72, 32 * wn + 16 * ni, kk, lane);
                sc[ni] = wmma_f16(a, bf, sc[ni]);
            }
        }
#pragma unroll
        for (int ni = 0; ni < 2; ++ni)
#pragma unroll
            for (int r = 0; r < 8; ++r)
                Sc[16 * wm + r + 8 * (lane >> 4)][32 * wn + 16 * ni + (lane & 15)] = sc[ni][r];
        __syncthreads();

        // online softmax stats (one thread per query row)
        if (tid < 64) {
            float mold = rmax[tid];
            float mnew = mold;
#pragma unroll 8
            for (int j = 0; j < 64; ++j) {
                float s = (smask[j] == 0) ? -3.0e38f : Sc[tid][j];
                mnew = fmaxf(mnew, s);
            }
            float scl = __expf(mold - mnew);
            float lsum = 0.0f;
#pragma unroll 8
            for (int j = 0; j < 64; ++j) {
                float p = 0.0f;
                if (smask[j] != 0) p = __expf(Sc[tid][j] - mnew);
                lsum += p;
                Ps[tid][j] = (_Float16)p;
            }
            rscl[tid] = scl;
            rmax[tid] = mnew;
            rsum[tid] = rsum[tid] * scl + lsum;
        }
        __syncthreads();

        // rescale running context, then O += P @ V_tile (contract over keys)
#pragma unroll
        for (int ni = 0; ni < 2; ++ni)
#pragma unroll
            for (int r = 0; r < 8; ++r)
                o[ni][r] *= rscl[16 * wm + r + 8 * (lane >> 4)];
#pragma unroll
        for (int kk = 0; kk < 64; kk += 32) {
            v16h a = load_frag_a(&Ps[0][0], 72, 16 * wm, kk, lane);
#pragma unroll
            for (int ni = 0; ni < 2; ++ni) {
                v16h bf = load_frag_b(&Vt[0][0], 72, 32 * wn + 16 * ni, kk, lane);
                o[ni] = wmma_f16(a, bf, o[ni]);
            }
        }
    }

    // normalize and store context f16 in (b, s, h*DH + d) layout
#pragma unroll
    for (int ni = 0; ni < 2; ++ni)
#pragma unroll
        for (int r = 0; r < 8; ++r) {
            int row = 16 * wm + r + 8 * (lane >> 4);
            int d   = 32 * wn + 16 * ni + (lane & 15);
            float inv = 1.0f / rsum[row];
            int s = qt * 64 + row;
            Ctx[((size_t)(b * SEQ + s)) * DIM + h * DH + d] = (_Float16)(o[ni][r] * inv);
        }
}

// ---------------------------------------------------------------------------
// Kernel 3: out = Ctx @ Wo^T + bo  (f16 A from workspace, f32 output)
// ---------------------------------------------------------------------------
__global__ __launch_bounds__(256)
void mhsa_outproj_kernel(const _Float16* __restrict__ Ctx, const float* __restrict__ W,
                         const float* __restrict__ bias, float* __restrict__ Out) {
    __shared__ _Float16 As[128][72];
    __shared__ _Float16 Bs[64][72];

    const int tid  = threadIdx.x;
    const int lane = tid & 31;
    const int w    = tid >> 5;
    const int wm   = w >> 1;
    const int wn   = w & 1;
    const int m_block = blockIdx.x * 128;
    const int n_block = blockIdx.y * 64;

    v8f acc[2][2] = {};

    const int ar = tid >> 1, ac = (tid & 1) * 32;
    const int br = tid >> 2, bc = (tid & 3) * 16;

    for (int k0 = 0; k0 < DIM; k0 += 64) {
        __syncthreads();
        {   // stage Ctx tile (already f16)
            const v8h* gp = (const v8h*)(Ctx + (size_t)(m_block + ar) * DIM + k0 + ac);
#pragma unroll
            for (int i = 0; i < 4; ++i)
                *(v8h*)&As[ar][ac + 8 * i] = gp[i];
        }
        {   // stage Wo tile (f32 -> f16)
            const float* gw = W + (size_t)(n_block + br) * DIM + k0 + bc;
#pragma unroll
            for (int i = 0; i < 2; ++i) {
                v4f f0 = *(const v4f*)(gw + 8 * i);
                v4f f1 = *(const v4f*)(gw + 8 * i + 4);
                *(v8h*)&Bs[br][bc + 8 * i] = cvt8(f0, f1);
            }
        }
        __syncthreads();
#pragma unroll
        for (int kk = 0; kk < 64; kk += 32) {
            v16h a0 = load_frag_a(&As[0][0], 72, 32 * wm,      kk, lane);
            v16h a1 = load_frag_a(&As[0][0], 72, 32 * wm + 16, kk, lane);
            v16h b0 = load_frag_b(&Bs[0][0], 72, 32 * wn,      kk, lane);
            v16h b1 = load_frag_b(&Bs[0][0], 72, 32 * wn + 16, kk, lane);
            acc[0][0] = wmma_f16(a0, b0, acc[0][0]);
            acc[0][1] = wmma_f16(a0, b1, acc[0][1]);
            acc[1][0] = wmma_f16(a1, b0, acc[1][0]);
            acc[1][1] = wmma_f16(a1, b1, acc[1][1]);
        }
    }

#pragma unroll
    for (int mi = 0; mi < 2; ++mi)
#pragma unroll
        for (int ni = 0; ni < 2; ++ni)
#pragma unroll
            for (int r = 0; r < 8; ++r) {
                int m = m_block + 32 * wm + 16 * mi + r + 8 * (lane >> 4);
                int n = n_block + 32 * wn + 16 * ni + (lane & 15);
                Out[(size_t)m * DIM + n] = acc[mi][ni][r] + bias[n];
            }
}

// ---------------------------------------------------------------------------
extern "C" void kernel_launch(void* const* d_in, const int* in_sizes, int n_in,
                              void* d_out, int out_size, void* d_ws, size_t ws_size,
                              hipStream_t stream) {
    (void)in_sizes; (void)n_in; (void)out_size; (void)ws_size;

    const float* q    = (const float*)d_in[0];
    const float* k    = (const float*)d_in[1];
    const float* v    = (const float*)d_in[2];
    const int*   mask = (const int*)d_in[3];
    const float* Wq = (const float*)d_in[4];  const float* bq = (const float*)d_in[5];
    const float* Wk = (const float*)d_in[6];  const float* bk = (const float*)d_in[7];
    const float* Wv = (const float*)d_in[8];  const float* bv = (const float*)d_in[9];
    const float* Wo = (const float*)d_in[10]; const float* bo = (const float*)d_in[11];
    float* out = (float*)d_out;

    const size_t NT = (size_t)BS * SEQ * DIM;   // 8,388,608 elements
    _Float16* Qh  = (_Float16*)d_ws;            // 16 MB each
    _Float16* Kh  = Qh + NT;
    _Float16* Vh  = Kh + NT;
    _Float16* Ctx = Vh + NT;                    // total 64 MB of workspace

    dim3 blk(256);
    dim3 gproj((BS * SEQ) / 128, DIM / 64);     // (64, 16)
    const float qscale = 0.125f;                // 1/sqrt(DH)

    mhsa_proj_kernel<<<gproj, blk, 0, stream>>>(q, Wq, bq, Qh, qscale);
    mhsa_proj_kernel<<<gproj, blk, 0, stream>>>(k, Wk, bk, Kh, 1.0f);
    mhsa_proj_kernel<<<gproj, blk, 0, stream>>>(v, Wv, bv, Vh, 1.0f);

    dim3 gattn(SEQ / 64, NH, BS);               // (32, 16, 4)
    mhsa_attn_kernel<<<gattn, blk, 0, stream>>>(Qh, Kh, Vh, mask, Ctx);

    mhsa_outproj_kernel<<<gproj, blk, 0, stream>>>(Ctx, Wo, bo, out);
}
